// MultiHeadAttention_2851858284782
// MI455X (gfx1250) — matrix operand order repaired and checked by reading, not compiled, not run on hardware
//
#include <hip/hip_runtime.h>
#include <cstdint>
#include <cstddef>

// ---------------------------------------------------------------------------
// MI455X (gfx1250) fused MHA block.
// All GEMMs run through v_wmma_f32_16x16x32_bf16 (fp32 -> bf16 convert while
// staging tiles into LDS).  wave32 everywhere.
// Round 2: software-pipelined K-loop (register double buffering) + explicit
// global_prefetch of tile t+2 so HBM latency hides behind the WMMAs.
// ---------------------------------------------------------------------------

typedef __bf16 bf16;
typedef __attribute__((ext_vector_type(8)))  bf16  v8bf;
typedef __attribute__((ext_vector_type(16))) bf16  v16bf;
typedef __attribute__((ext_vector_type(8)))  float v8f;

// ---------------------------------------------------------------------------
// Generic tiled GEMM:  C[z][m][n] = sum_k A[z][m][k] * B'[k][n]  (+ bias[n])
//   BT = true  : B' = Bsrc^T, i.e. Bsrc is [N][K] row-major (X @ W^T, Q @ K^T)
//   BT = false : B' = Bsrc,   i.e. Bsrc is [K][N] row-major (P @ V)
// z decomposes as (b = z / Hn, h = z % Hn).
// Requires: gridDim.y*128 == M, gridDim.x*64 == N, K % 32 == 0 (all true here).
//
// Block: 256 threads = 8 waves, 4x2 wave grid, 32x32 C tile per wave
// (2x2 accumulators of v_wmma_f32_16x16x32_bf16).
// LDS pitch 40 bf16 = 80 B => every v8bf fragment load is 16B-aligned.
// ---------------------------------------------------------------------------
template <bool BT>
__global__ __launch_bounds__(256) void gemm_bf16_wmma(
    const float* __restrict__ A, int lda, long long sAb, long long sAh,
    const float* __restrict__ Bsrc, int ldb, long long sBb, long long sBh,
    float* __restrict__ C, int ldc, long long sCb, long long sCh,
    int K, int Hn, const float* __restrict__ bias)
{
    __shared__ bf16 lA[128][40];   // [m][k], pitch 80B
    __shared__ bf16 lB[64][40];    // [n][k], pitch 80B

    const int tid   = threadIdx.x;
    const int lane  = tid & 31;
    const int w     = tid >> 5;
    const int wm    = (w & 3) * 32;     // wave row offset within block tile
    const int wn    = (w >> 2) * 32;    // wave col offset within block tile
    const int m0    = blockIdx.y * 128;
    const int n0    = blockIdx.x * 64;
    const int zb    = (int)blockIdx.z / Hn;
    const int zh    = (int)blockIdx.z % Hn;

    const float* Ab = A    + zb * sAb + zh * sAh;
    const float* Bb = Bsrc + zb * sBb + zh * sBh;
    float*       Cb = C    + zb * sCb + zh * sCh;

    const int row16 = lane & 15;   // m (A frag) / n (B frag) within 16
    const int half  = lane >> 4;   // lane-half selects K sub-range

    v8f acc[2][2];
#pragma unroll
    for (int i = 0; i < 2; ++i)
#pragma unroll
        for (int j = 0; j < 2; ++j) acc[i][j] = {};

    // per-thread staging registers (tile t+1 lives here while tile t computes)
    float4 ra[4];
    float4 rb[2];

    // ---- global -> registers for K-tile at k0
    auto gload = [&](int k0) {
#pragma unroll
        for (int i = 0; i < 4; ++i) {
            const int idx = tid + i * 256;           // 0..1023
            const int r   = idx >> 3;
            const int c   = (idx & 7) * 4;
            ra[i] = *reinterpret_cast<const float4*>(
                Ab + (size_t)(m0 + r) * lda + (k0 + c));
        }
        if (BT) {
#pragma unroll
            for (int i = 0; i < 2; ++i) {
                const int idx = tid + i * 256;       // 0..511
                const int r   = idx >> 3;            // n 0..63
                const int c   = (idx & 7) * 4;       // k
                rb[i] = *reinterpret_cast<const float4*>(
                    Bb + (size_t)(n0 + r) * ldb + (k0 + c));
            }
        } else {
#pragma unroll
            for (int i = 0; i < 2; ++i) {
                const int idx = tid + i * 256;       // 0..511
                const int r   = idx >> 4;            // k 0..31
                const int c   = (idx & 15) * 4;      // n
                rb[i] = *reinterpret_cast<const float4*>(
                    Bb + (size_t)(k0 + r) * ldb + (n0 + c));
            }
        }
    };

    // ---- registers -> LDS (fp32 -> bf16 convert), layout lA[m][k], lB[n][k]
    auto lstore = [&]() {
#pragma unroll
        for (int i = 0; i < 4; ++i) {
            const int idx = tid + i * 256;
            const int r   = idx >> 3;
            const int c   = (idx & 7) * 4;
            lA[r][c + 0] = (bf16)ra[i].x; lA[r][c + 1] = (bf16)ra[i].y;
            lA[r][c + 2] = (bf16)ra[i].z; lA[r][c + 3] = (bf16)ra[i].w;
        }
        if (BT) {
#pragma unroll
            for (int i = 0; i < 2; ++i) {
                const int idx = tid + i * 256;
                const int r   = idx >> 3;
                const int c   = (idx & 7) * 4;
                lB[r][c + 0] = (bf16)rb[i].x; lB[r][c + 1] = (bf16)rb[i].y;
                lB[r][c + 2] = (bf16)rb[i].z; lB[r][c + 3] = (bf16)rb[i].w;
            }
        } else {
#pragma unroll
            for (int i = 0; i < 2; ++i) {
                const int idx = tid + i * 256;
                const int r   = idx >> 4;            // k
                const int c   = (idx & 15) * 4;      // n
                lB[c + 0][r] = (bf16)rb[i].x; lB[c + 1][r] = (bf16)rb[i].y;
                lB[c + 2][r] = (bf16)rb[i].z; lB[c + 3][r] = (bf16)rb[i].w;
            }
        }
    };

    // ---- L2 prefetch of K-tile at k0 (emits global_prefetch_b8)
    auto pf = [&](int k0) {
#pragma unroll
        for (int i = 0; i < 4; ++i) {
            const int idx = tid + i * 256;
            const int r   = idx >> 3;
            const int c   = (idx & 7) * 4;
            __builtin_prefetch(Ab + (size_t)(m0 + r) * lda + (k0 + c), 0, 3);
        }
#pragma unroll
        for (int i = 0; i < 2; ++i) {
            const int idx = tid + i * 256;
            if (BT) {
                const int r = idx >> 3, c = (idx & 7) * 4;
                __builtin_prefetch(Bb + (size_t)(n0 + r) * ldb + (k0 + c), 0, 3);
            } else {
                const int r = idx >> 4, c = (idx & 15) * 4;
                __builtin_prefetch(Bb + (size_t)(k0 + r) * ldb + (n0 + c), 0, 3);
            }
        }
    };

    gload(0);                      // prologue: tile 0 into registers
    if (K > 32) pf(32);            // and warm L2 for tile 1

    for (int k0 = 0; k0 < K; k0 += 32) {
        lstore();                  // stage tile t (regs -> LDS, cvt to bf16)
        __syncthreads();

        const int kn = k0 + 32;
        if (kn < K)      gload(kn);     // overlap: tile t+1 global loads
        if (kn + 32 < K) pf(kn + 32);   // overlap: tile t+2 L2 prefetch

        // ---- fragments (ISA 7.12.2 16-bit layouts) + WMMAs on tile t
        v16bf afrag[2], bfrag[2];
#pragma unroll
        for (int mi = 0; mi < 2; ++mi) {
            const bf16* base = &lA[wm + mi * 16 + row16][half * 8];
            const v8bf lo = *reinterpret_cast<const v8bf*>(base);        // K = h*8 ..
            const v8bf hi = *reinterpret_cast<const v8bf*>(base + 16);   // K = 16+h*8 ..
            afrag[mi] = __builtin_shufflevector(lo, hi,
                0, 1, 2, 3, 4, 5, 6, 7, 8, 9, 10, 11, 12, 13, 14, 15);
        }
#pragma unroll
        for (int ni = 0; ni < 2; ++ni) {
            const bf16* base = &lB[wn + ni * 16 + row16][half * 8];
            const v8bf lo = *reinterpret_cast<const v8bf*>(base);        // K = h*8 ..
            const v8bf hi = *reinterpret_cast<const v8bf*>(base + 16);   // K = 16+h*8 ..
            bfrag[ni] = __builtin_shufflevector(lo, hi,
                0, 1, 2, 3, 4, 5, 6, 7, 8, 9, 10, 11, 12, 13, 14, 15);
        }
#pragma unroll
        for (int mi = 0; mi < 2; ++mi)
#pragma unroll
            for (int ni = 0; ni < 2; ++ni)
                acc[mi][ni] = __builtin_amdgcn_wmma_f32_16x16x32_bf16(
                    false, afrag[mi], false, bfrag[ni],
                    (short)0, acc[mi][ni], false, false);
        __syncthreads();
    }

    // ---- epilogue: C layout (VGPR r: lanes 0-15 -> M=r, lanes 16-31 -> M=8+r)
#pragma unroll
    for (int mi = 0; mi < 2; ++mi) {
        const int mbase = m0 + wm + mi * 16 + half * 8;
#pragma unroll
        for (int ni = 0; ni < 2; ++ni) {
            const int n  = n0 + wn + ni * 16 + row16;
            const float bv = bias ? bias[n] : 0.0f;
#pragma unroll
            for (int r = 0; r < 8; ++r)
                Cb[(size_t)(mbase + r) * ldc + n] = acc[mi][ni][r] + bv;
        }
    }
}

// ---------------------------------------------------------------------------
// In-place masked softmax over rows of length S (=2048).  One 256-thread
// block (8 wave32) per row; scale = 1/sqrt(HEAD_DIM) fused here so the
// scores GEMM stores raw Q.K^T.
// ---------------------------------------------------------------------------
__global__ __launch_bounds__(256) void softmax_mask(
    float* __restrict__ probs, const int* __restrict__ mask,
    int S, int Hn, float scale)
{
    const int row = blockIdx.x;             // 0 .. B*H*S-1
    const int b   = row / (Hn * S);
    const int qi  = row % S;
    float*       p    = probs + (size_t)row * S;
    const int*   mrow = mask + ((size_t)b * S + qi) * S;
    const int    tid  = threadIdx.x;

    float vals[8];
    float mx = -3.4e38f;
#pragma unroll
    for (int i = 0; i < 8; ++i) {
        const int j = tid + i * 256;
        const float vv = p[j] * scale + (float)(1 - mrow[j]) * -1.0e9f;
        vals[i] = vv;
        mx = fmaxf(mx, vv);
    }

    __shared__ float red[8];
#pragma unroll
    for (int off = 16; off > 0; off >>= 1)
        mx = fmaxf(mx, __shfl_xor(mx, off, 32));
    if ((tid & 31) == 0) red[tid >> 5] = mx;
    __syncthreads();
    if (tid == 0) {
        float m = red[0];
        for (int i = 1; i < 8; ++i) m = fmaxf(m, red[i]);
        red[0] = m;
    }
    __syncthreads();
    mx = red[0];
    __syncthreads();

    float s = 0.0f;
#pragma unroll
    for (int i = 0; i < 8; ++i) { vals[i] = __expf(vals[i] - mx); s += vals[i]; }
#pragma unroll
    for (int off = 16; off > 0; off >>= 1) s += __shfl_xor(s, off, 32);
    if ((tid & 31) == 0) red[tid >> 5] = s;
    __syncthreads();
    if (tid == 0) {
        float t = 0.0f;
        for (int i = 0; i < 8; ++i) t += red[i];
        red[0] = t;
    }
    __syncthreads();
    const float inv = 1.0f / red[0];
#pragma unroll
    for (int i = 0; i < 8; ++i) p[tid + i * 256] = vals[i] * inv;
}

// ---------------------------------------------------------------------------
// out = LayerNorm(residual + x) * gamma + beta.  One block per row of D=1024.
// ---------------------------------------------------------------------------
__global__ __launch_bounds__(256) void layernorm_residual(
    const float* __restrict__ resid, const float* __restrict__ xin,
    const float* __restrict__ gamma, const float* __restrict__ beta,
    float* __restrict__ out, int D)
{
    const int row = blockIdx.x;
    const int tid = threadIdx.x;
    const float* rr = resid + (size_t)row * D;
    const float* xx = xin   + (size_t)row * D;

    float x[4];
    float s = 0.0f, s2 = 0.0f;
#pragma unroll
    for (int i = 0; i < 4; ++i) {
        const int j = tid + i * 256;
        const float vv = rr[j] + xx[j];
        x[i] = vv; s += vv; s2 += vv * vv;
    }
#pragma unroll
    for (int off = 16; off > 0; off >>= 1) {
        s  += __shfl_xor(s,  off, 32);
        s2 += __shfl_xor(s2, off, 32);
    }
    __shared__ float rs[8], rs2[8];
    if ((tid & 31) == 0) { rs[tid >> 5] = s; rs2[tid >> 5] = s2; }
    __syncthreads();
    if (tid == 0) {
        float a = 0.0f, b = 0.0f;
        for (int i = 0; i < 8; ++i) { a += rs[i]; b += rs2[i]; }
        rs[0] = a; rs2[0] = b;
    }
    __syncthreads();
    const float mu   = rs[0] / (float)D;
    const float var  = rs2[0] / (float)D - mu * mu;
    const float rstd = rsqrtf(var + 1e-5f);
#pragma unroll
    for (int i = 0; i < 4; ++i) {
        const int j = tid + i * 256;
        out[(size_t)row * D + j] = (x[i] - mu) * rstd * gamma[j] + beta[j];
    }
}

// ---------------------------------------------------------------------------
// Host orchestration.  Workspace (48 MB):
//   ws[0 ..)        Qproj  (later reused as attn_out)
//   ws[4M ..)       Kproj  (later reused as attn_proj = attn_out @ Wo^T + bo)
//   ws[8M ..)       Vproj
// d_out = [ outputs (B*S*D) | attn_probs (B*H*S*S) ]; scores are written into
// the probs region and softmaxed in place.
// ---------------------------------------------------------------------------
extern "C" void kernel_launch(void* const* d_in, const int* in_sizes, int n_in,
                              void* d_out, int out_size, void* d_ws, size_t ws_size,
                              hipStream_t stream)
{
    (void)in_sizes; (void)n_in; (void)out_size; (void)ws_size;

    const float* q     = (const float*)d_in[0];
    const float* k     = (const float*)d_in[1];
    const float* v     = (const float*)d_in[2];
    const int*   mask  = (const int*)  d_in[3];
    const float* Wq    = (const float*)d_in[4];
    const float* Wk    = (const float*)d_in[5];
    const float* Wv    = (const float*)d_in[6];
    const float* Wo    = (const float*)d_in[7];
    const float* bo    = (const float*)d_in[8];
    const float* gamma = (const float*)d_in[9];
    const float* beta  = (const float*)d_in[10];

    const int Bn = 2, S = 2048, D = 1024, H = 16, HD = 64;
    const size_t BS = (size_t)Bn * S;            // 4096 rows

    float* Qp = (float*)d_ws;                    // [BS][D]  (reused: attn_out)
    float* Kp = Qp + BS * D;                     // [BS][D]  (reused: attn_proj)
    float* Vp = Kp + BS * D;                     // [BS][D]
    float* outp  = (float*)d_out;                // [BS][D]
    float* probs = outp + BS * D;                // [B][H][S][S]

    const dim3 blk(256);
    const dim3 gproj(D / 64, (unsigned)(BS / 128), 1);     // (16, 32, 1)

    // --- input projections: X @ W^T  (M=4096, N=1024, K=1024)
    gemm_bf16_wmma<true><<<gproj, blk, 0, stream>>>(
        q, D, 0, 0,  Wq, D, 0, 0,  Qp, D, 0, 0,  D, 1, nullptr);
    gemm_bf16_wmma<true><<<gproj, blk, 0, stream>>>(
        k, D, 0, 0,  Wk, D, 0, 0,  Kp, D, 0, 0,  D, 1, nullptr);
    gemm_bf16_wmma<true><<<gproj, blk, 0, stream>>>(
        v, D, 0, 0,  Wv, D, 0, 0,  Vp, D, 0, 0,  D, 1, nullptr);

    // --- scores: per (b,h)  Qh @ Kh^T  (M=N=2048, K=64) -> probs region
    const dim3 gsc(S / 64, S / 128, Bn * H);               // (32, 16, 32)
    gemm_bf16_wmma<true><<<gsc, blk, 0, stream>>>(
        Qp, D, (long long)S * D, HD,
        Kp, D, (long long)S * D, HD,
        probs, S, (long long)H * S * S, (long long)S * S,
        HD, H, nullptr);

    // --- masked softmax in place (scale = 1/sqrt(64))
    softmax_mask<<<dim3((unsigned)(Bn * H * S)), blk, 0, stream>>>(
        probs, mask, S, H, 0.125f);

    // --- attn_out: per (b,h)  P @ Vh  (M=2048, N=64, K=2048) -> Qp (reused)
    const dim3 gao(HD / 64, S / 128, Bn * H);              // (1, 16, 32)
    gemm_bf16_wmma<false><<<gao, blk, 0, stream>>>(
        probs, S, (long long)H * S * S, (long long)S * S,
        Vp, D, (long long)S * D, HD,
        Qp, D, (long long)S * D, HD,
        S, H, nullptr);

    // --- output projection: attn_out @ Wo^T + bo -> Kp (reused)
    gemm_bf16_wmma<true><<<gproj, blk, 0, stream>>>(
        Qp, D, 0, 0,  Wo, D, 0, 0,  Kp, D, 0, 0,  D, 1, bo);

    // --- residual + LayerNorm -> outputs
    layernorm_residual<<<dim3((unsigned)BS), blk, 0, stream>>>(
        q, Kp, gamma, beta, outp, D);
}
